// gMAE_41781441855574
// MI455X (gfx1250) — compile-verified, matrix operand order checked by reading, not run on hardware
//
#include <hip/hip_runtime.h>
#include <hip/hip_fp16.h>
#include <math.h>

// ---- problem constants (from reference) ----
#define BB 8
#define LL 64
#define DD 16
#define HID 512
#define TEc 16
#define PP 8
#define EE 64            // HID/P
#define NHEAD 8
#define HD 8             // E/NHEAD
#define MLEN 4
#define NVIS 4
#define SS 128           // B*D
#define FINc 17          // 1+TE
#define FPAD 32          // FIN padded to one WMMA K-step
#define NLAYERS 4
#define SL 8192          // S*L
#define CTOT (FINc*HID)  // 8704 channels in fg3 output

typedef __attribute__((ext_vector_type(16))) _Float16 v16h;
typedef __attribute__((ext_vector_type(8)))  _Float16 v8h;
typedef __attribute__((ext_vector_type(8)))  float    v8f;

// ------------------------------------------------------------------
// WMMA helpers (gfx1250, wave32). 16-bit A-matrix 16x32 layout:
//  lanes 0-15 : M=lane,     K in {k0..k0+7} U {k0+16..k0+23}
//  lanes 16-31: M=lane-16,  K in {k0+8..k0+15} U {k0+24..k0+31}
// B (32x16) is loaded with the same per-lane pattern from the row-major
// weight matrix W(N,K): B[k][n] = W[n][k].
// ------------------------------------------------------------------
__device__ __forceinline__ v16h load_frag(const _Float16* base, int row0, int ld,
                                          int k0, int lane) {
  const _Float16* p = base + (size_t)(row0 + (lane & 15)) * ld
                           + (size_t)(k0 + ((lane >> 4) << 3));
  v8h lo = *(const v8h*)(p);
  v8h hi = *(const v8h*)(p + 16);
  return __builtin_shufflevector(lo, hi, 0,1,2,3,4,5,6,7,8,9,10,11,12,13,14,15);
}

__device__ __forceinline__ v8f wmma16(v16h a, v16h b, v8f c) {
  return __builtin_amdgcn_wmma_f32_16x16x32_f16(false, a, false, b, (short)0, c,
                                                false, false);
}

// ------------------------------------------------------------------
// small utility kernels
// ------------------------------------------------------------------
__global__ void f32_to_f16_kernel(const float* __restrict__ src,
                                  _Float16* __restrict__ dst, int n) {
  int i = blockIdx.x * blockDim.x + threadIdx.x;
  if (i < n) dst[i] = (_Float16)src[i];
}

__global__ void fg1_pad_kernel(const float* __restrict__ w,
                               _Float16* __restrict__ dst) {
  int i = blockIdx.x * blockDim.x + threadIdx.x;   // 512*32
  if (i >= HID * FPAD) return;
  int n = i >> 5, k = i & 31;
  dst[i] = (_Float16)((k < FINc) ? w[n * FINc + k] : 0.f);
}

// build Xf (S*L,17) in f32 and zero-padded f16 (S*L,32), plus mask m(S*L)
__global__ void build_xf_kernel(const float* __restrict__ value,
                                const float* __restrict__ tts,
                                const float* __restrict__ mask,
                                const float* __restrict__ tsw,
                                const float* __restrict__ tsb,
                                const float* __restrict__ tpw,
                                const float* __restrict__ tpb,
                                float* __restrict__ xf32,
                                _Float16* __restrict__ xf16,
                                float* __restrict__ mbuf) {
  int idx = blockIdx.x * blockDim.x + threadIdx.x;
  if (idx >= SL) return;
  int s = idx / LL, l = idx % LL;
  int b = s / DD, d = s % DD;
  float X  = value[(b * LL + l) * DD + d];
  float tt = tts[b * LL + l];
  float m  = mask[(b * LL + l) * DD + d];
  float f[FINc];
  f[0] = X;
  f[1] = tt * tsw[0] + tsb[0];
  #pragma unroll
  for (int j = 0; j < TEc - 1; ++j) f[2 + j] = __sinf(tt * tpw[j] + tpb[j]);
  #pragma unroll
  for (int j = 0; j < FINc; ++j) xf32[idx * FINc + j] = f[j];
  #pragma unroll
  for (int j = 0; j < FINc; ++j) xf16[idx * FPAD + j] = (_Float16)f[j];
  #pragma unroll
  for (int j = FINc; j < FPAD; ++j) xf16[idx * FPAD + j] = (_Float16)0.f;
  mbuf[idx] = m;
}

// stable argsort of 8 noise values per row; mask_idx = perm[:4] (perm order),
// vis_idx = sort(perm[4:]) ascending
__global__ void sort_kernel(const float* __restrict__ noise,
                            int* __restrict__ vis_idx,
                            int* __restrict__ mask_idx) {
  int s = blockIdx.x * blockDim.x + threadIdx.x;
  if (s >= SS) return;
  float v[PP]; int idx[PP];
  #pragma unroll
  for (int p = 0; p < PP; ++p) { v[p] = noise[s * PP + p]; idx[p] = p; }
  for (int i = 1; i < PP; ++i) {
    float kv = v[i]; int ki = idx[i]; int j = i - 1;
    while (j >= 0 && v[j] > kv) { v[j+1] = v[j]; idx[j+1] = idx[j]; --j; }
    v[j+1] = kv; idx[j+1] = ki;
  }
  #pragma unroll
  for (int p = 0; p < MLEN; ++p) mask_idx[s * MLEN + p] = idx[p];
  int vi[NVIS];
  #pragma unroll
  for (int p = 0; p < NVIS; ++p) vi[p] = idx[MLEN + p];
  for (int i = 1; i < NVIS; ++i) {
    int kv = vi[i]; int j = i - 1;
    while (j >= 0 && vi[j] > kv) { vi[j+1] = vi[j]; --j; }
    vi[j+1] = kv;
  }
  #pragma unroll
  for (int p = 0; p < NVIS; ++p) vis_idx[s * NVIS + p] = vi[p];
}

// ------------------------------------------------------------------
// WMMA GEMM: C = act(A(MxK,f16) @ W(NxK,f16)^T + bias).
// 2x2 register tile per wave (32x32 output): 4 WMMAs per 4 fragment
// loads -> 2x the arithmetic intensity of the naive 1-tile version.
// M and N must be multiples of 32 (true for all call sites here).
// ------------------------------------------------------------------
__global__ __launch_bounds__(128) void gemm_wmma_kernel(
    const _Float16* __restrict__ A, const _Float16* __restrict__ W,
    const float* __restrict__ bias, _Float16* __restrict__ C16,
    float* __restrict__ C32, int M, int N, int K, int do_relu) {
  int lane = threadIdx.x & 31;
  int wid  = threadIdx.x >> 5;
  int tile = blockIdx.x * (blockDim.x >> 5) + wid;
  int ntn  = N >> 5;
  if (tile >= (M >> 5) * ntn) return;
  int m0 = (tile / ntn) << 5;
  int n0 = (tile % ntn) << 5;
  v8f acc00 = {0.f,0.f,0.f,0.f,0.f,0.f,0.f,0.f};
  v8f acc01 = acc00, acc10 = acc00, acc11 = acc00;
  for (int k0 = 0; k0 < K; k0 += 32) {
    if (k0 + 32 < K) {  // pull next K-slab toward L2/L0 (global_prefetch_b8)
      __builtin_prefetch(A + (size_t)(m0 + (lane & 31)) * K + k0 + 32, 0, 1);
      __builtin_prefetch(W + (size_t)(n0 + (lane & 31)) * K + k0 + 32, 0, 1);
    }
    v16h a0 = load_frag(A, m0,      K, k0, lane);
    v16h a1 = load_frag(A, m0 + 16, K, k0, lane);
    v16h b0 = load_frag(W, n0,      K, k0, lane);
    v16h b1 = load_frag(W, n0 + 16, K, k0, lane);
    acc00 = wmma16(a0, b0, acc00);
    acc01 = wmma16(a0, b1, acc01);
    acc10 = wmma16(a1, b0, acc10);
    acc11 = wmma16(a1, b1, acc11);
  }
  int nc = lane & 15;
  int mr = (lane >> 4) << 3;
  float bi0 = bias ? bias[n0 + nc]      : 0.f;
  float bi1 = bias ? bias[n0 + 16 + nc] : 0.f;
  #pragma unroll
  for (int r = 0; r < 8; ++r) {
    int ma = m0 + r + mr;
    int mb = m0 + 16 + r + mr;
    float v00 = acc00[r] + bi0, v01 = acc01[r] + bi1;
    float v10 = acc10[r] + bi0, v11 = acc11[r] + bi1;
    if (do_relu) {
      v00 = fmaxf(v00, 0.f); v01 = fmaxf(v01, 0.f);
      v10 = fmaxf(v10, 0.f); v11 = fmaxf(v11, 0.f);
    }
    if (C16) {
      C16[(size_t)ma * N + n0 + nc]      = (_Float16)v00;
      C16[(size_t)ma * N + n0 + 16 + nc] = (_Float16)v01;
      C16[(size_t)mb * N + n0 + nc]      = (_Float16)v10;
      C16[(size_t)mb * N + n0 + 16 + nc] = (_Float16)v11;
    }
    if (C32) {
      C32[(size_t)ma * N + n0 + nc]      = v00;
      C32[(size_t)ma * N + n0 + 16 + nc] = v01;
      C32[(size_t)mb * N + n0 + nc]      = v10;
      C32[(size_t)mb * N + n0 + 16 + nc] = v11;
    }
  }
}

// ------------------------------------------------------------------
// Fused fg3 GEMM + masked softmax over L + einsum('lf,lhf->h') per sample.
// One block per s. Waves stream 16-wide channel tiles of the (64 x 8704)
// logit matrix; `filt` (285 MB in f32) is never materialized. B fragment
// is reused across the 4 M-tiles (4 WMMAs per 5 fragment loads).
// ------------------------------------------------------------------
__global__ __launch_bounds__(256) void fg3_fused_kernel(
    const _Float16* __restrict__ H,   // (8192,512) f16 (post-fg2 relu)
    const _Float16* __restrict__ W3,  // (8704,512) f16
    const float* __restrict__ b3,     // (8704,)
    const float* __restrict__ Xf,     // (8192,17) f32
    const float* __restrict__ mbuf,   // (8192,)
    const float* __restrict__ t_bias, // (512,)
    float* __restrict__ outp)         // (128,512) == `output` section of d_out
{
  __shared__ float Xfs[LL * FINc];
  __shared__ float ms[LL];
  __shared__ float ttcn[HID];
  __shared__ float ltile[8][LL * 16];

  int s = blockIdx.x;
  int tid = threadIdx.x;
  int lane = tid & 31, wid = tid >> 5;
  const _Float16* Hbase = H + (size_t)s * LL * HID;

  for (int i = tid; i < LL * FINc; i += 256) Xfs[i] = Xf[(size_t)s * LL * FINc + i];
  if (tid < LL) ms[tid] = mbuf[s * LL + tid];
  for (int i = tid; i < HID; i += 256) ttcn[i] = 0.f;
  __syncthreads();

  const int NT = CTOT / 16;  // 544 channel tiles; 68 per wave, no tail
  for (int ct = wid; ct < NT; ct += 8) {
    int c0 = ct << 4;
    v8f a0c = {0.f,0.f,0.f,0.f,0.f,0.f,0.f,0.f};
    v8f a1c = a0c, a2c = a0c, a3c = a0c;
    for (int k0 = 0; k0 < HID; k0 += 32) {
      v16h bf = load_frag(W3, c0, HID, k0, lane);
      v16h f0 = load_frag(Hbase,  0, HID, k0, lane);
      v16h f1 = load_frag(Hbase, 16, HID, k0, lane);
      v16h f2 = load_frag(Hbase, 32, HID, k0, lane);
      v16h f3 = load_frag(Hbase, 48, HID, k0, lane);
      a0c = wmma16(f0, bf, a0c);
      a1c = wmma16(f1, bf, a1c);
      a2c = wmma16(f2, bf, a2c);
      a3c = wmma16(f3, bf, a3c);
    }
    int n = lane & 15;
    float bi = b3[c0 + n];
    float* lt = ltile[wid];
    #pragma unroll
    for (int r = 0; r < 8; ++r) {
      int mr = r + ((lane >> 4) << 3);
      lt[(mr +  0) * 16 + n] = a0c[r] + bi;
      lt[(mr + 16) * 16 + n] = a1c[r] + bi;
      lt[(mr + 32) * 16 + n] = a2c[r] + bi;
      lt[(mr + 48) * 16 + n] = a3c[r] + bi;
    }
    asm volatile("" ::: "memory");  // LDS ops are in-order per wave
    if (lane < 16) {                // one lane per channel column
      int c = c0 + lane;
      int h = c / FINc, f = c % FINc;
      float mx = -3.4e38f;
      for (int l = 0; l < LL; ++l) {
        float mm = ms[l];
        float z = lt[l * 16 + lane] * mm + (1.f - mm) * (-1e8f);
        lt[l * 16 + lane] = z;
        mx = fmaxf(mx, z);
      }
      float se = 0.f, wsum = 0.f;
      for (int l = 0; l < LL; ++l) {
        float e = __expf(lt[l * 16 + lane] - mx);
        se += e;
        wsum += e * Xfs[l * FINc + f];
      }
      atomicAdd(&ttcn[h], wsum / se);  // ds_add_f32
    }
    asm volatile("" ::: "memory");
  }
  __syncthreads();
  for (int h = tid; h < HID; h += 256) {
    float v = ttcn[h] + t_bias[h];
    outp[(size_t)s * HID + h] = fmaxf(v, 0.f);
  }
}

// ------------------------------------------------------------------
// small dense: Y(M,N) = act(X(M,K) @ W(N,K)^T + b); act 0=none 1=relu 2=gelu
// ------------------------------------------------------------------
__global__ void dense_kernel(const float* __restrict__ X, const float* __restrict__ W,
                             const float* __restrict__ b, float* __restrict__ Y,
                             int M, int N, int K, int act) {
  int idx = blockIdx.x * blockDim.x + threadIdx.x;
  if (idx >= M * N) return;
  int n = idx % N, m = idx / N;
  const float* x = X + (size_t)m * K;
  const float* w = W + (size_t)n * K;
  float acc = b ? b[n] : 0.f;
  for (int k = 0; k < K; ++k) acc += x[k] * w[k];
  if (act == 1) acc = fmaxf(acc, 0.f);
  else if (act == 2) acc = 0.5f * acc * (1.f + erff(acc * 0.70710678118654752440f));
  Y[idx] = acc;
}

// attention over all 512 rows, online softmax, one thread per (head, query)
__global__ void attn_kernel(const float* __restrict__ qkv, float* __restrict__ o) {
  int idx = blockIdx.x * blockDim.x + threadIdx.x;
  if (idx >= NHEAD * 512) return;
  int h = idx / 512, n = idx % 512;
  const float* q = qkv + (size_t)n * (3 * EE) + h * HD;
  float mx = -3.4e38f, se = 0.f;
  float acc[HD];
  #pragma unroll
  for (int d = 0; d < HD; ++d) acc[d] = 0.f;
  for (int m = 0; m < 512; ++m) {
    const float* k = qkv + (size_t)m * (3 * EE) + EE + h * HD;
    const float* v = qkv + (size_t)m * (3 * EE) + 2 * EE + h * HD;
    float sdot = 0.f;
    #pragma unroll
    for (int d = 0; d < HD; ++d) sdot += q[d] * k[d];
    sdot *= 0.35355339059327379f;  // HD^-0.5
    float nmx = fmaxf(mx, sdot);
    float scale = __expf(mx - nmx);
    float e = __expf(sdot - nmx);
    se = se * scale + e;
    #pragma unroll
    for (int d = 0; d < HD; ++d) acc[d] = acc[d] * scale + e * v[d];
    mx = nmx;
  }
  #pragma unroll
  for (int d = 0; d < HD; ++d) o[(size_t)n * EE + h * HD + d] = acc[d] / se;
}

// Y[n,:] = LN( X[n,:] + O(n,K) @ W(64,K)^T + b ) * g + beta  (64 threads/row)
__global__ __launch_bounds__(64) void addproj_ln_kernel(
    const float* __restrict__ X, const float* __restrict__ O,
    const float* __restrict__ W, const float* __restrict__ b,
    const float* __restrict__ g, const float* __restrict__ beta,
    float* __restrict__ Y, int K) {
  __shared__ float r1[EE], r2[EE];
  int n = blockIdx.x, j = threadIdx.x;
  const float* o = O + (size_t)n * K;
  const float* w = W + (size_t)j * K;
  float acc = b[j];
  for (int k = 0; k < K; ++k) acc += o[k] * w[k];
  float t = X[(size_t)n * EE + j] + acc;
  r1[j] = t; r2[j] = t * t;
  __syncthreads();
  for (int st = 32; st > 0; st >>= 1) {
    if (j < st) { r1[j] += r1[j + st]; r2[j] += r2[j + st]; }
    __syncthreads();
  }
  float mu = r1[0] * (1.f / EE);
  float var = r2[0] * (1.f / EE) - mu * mu;
  Y[(size_t)n * EE + j] = (t - mu) * rsqrtf(var + 1e-5f) * g[j] + beta[j];
}

__global__ void gather_vis_kernel(const float* __restrict__ outp,
                                  const int* __restrict__ vis_idx,
                                  float* __restrict__ xb) {
  int idx = blockIdx.x * blockDim.x + threadIdx.x;  // S*NVIS*E = 32768
  if (idx >= SS * NVIS * EE) return;
  int j = idx % EE, v = (idx / EE) % NVIS, s = idx / (EE * NVIS);
  xb[idx] = outp[(size_t)s * HID + vis_idx[s * NVIS + v] * EE + j];
}

__global__ void avg4_kernel(const float* __restrict__ x, float* __restrict__ z) {
  int idx = blockIdx.x * blockDim.x + threadIdx.x;  // S*E = 8192
  if (idx >= SS * EE) return;
  int s = idx / EE, j = idx % EE;
  z[idx] = 0.25f * (x[(s * 4 + 0) * EE + j] + x[(s * 4 + 1) * EE + j] +
                    x[(s * 4 + 2) * EE + j] + x[(s * 4 + 3) * EE + j]);
}

__global__ void bcast_full_kernel(const float* __restrict__ advb,
                                  float* __restrict__ full) {
  int idx = blockIdx.x * blockDim.x + threadIdx.x;  // S*P*E = 65536
  if (idx >= SS * PP * EE) return;
  int j = idx % EE, s = idx / (PP * EE);
  full[idx] = advb[s * EE + j];
}

__global__ void scatter_vis_kernel(const float* __restrict__ dvb,
                                   const int* __restrict__ vis_idx,
                                   float* __restrict__ full) {
  int idx = blockIdx.x * blockDim.x + threadIdx.x;  // S*NVIS*E = 32768
  if (idx >= SS * NVIS * EE) return;
  int j = idx % EE, v = (idx / EE) % NVIS, s = idx / (EE * NVIS);
  int p = vis_idx[s * NVIS + v];
  full[(size_t)s * HID + p * EE + j] = dvb[(s * NVIS + v) * EE + j];
}

// sig = full(S,512) @ sp_w(64,512)^T + sp_b, scattered to (B,L,D); sqdiff out
__global__ void sig_kernel(const float* __restrict__ full,
                           const float* __restrict__ spw,
                           const float* __restrict__ spb,
                           const float* __restrict__ value,
                           float* __restrict__ sig_out,
                           float* __restrict__ sq_sig) {
  int idx = blockIdx.x * blockDim.x + threadIdx.x;  // S*L = 8192
  if (idx >= SS * LL) return;
  int s = idx / LL, l = idx % LL;
  const float* f = full + (size_t)s * HID;
  const float* w = spw + (size_t)l * HID;
  float a = spb[l];
  for (int k = 0; k < HID; ++k) a += f[k] * w[k];
  int b = s / DD, d = s % DD;
  sig_out[(b * LL + l) * DD + d] = a;
  float diff = a - value[(b * LL + l) * DD + d];
  sq_sig[idx] = diff * diff;
}

__global__ void latent_kernel(const float* __restrict__ full,
                              const float* __restrict__ tokens,
                              const int* __restrict__ mask_idx,
                              float* __restrict__ sq_lat) {
  int idx = blockIdx.x * blockDim.x + threadIdx.x;  // S*MLEN*E = 32768
  if (idx >= SS * MLEN * EE) return;
  int j = idx % EE, mi = (idx / EE) % MLEN, s = idx / (EE * MLEN);
  int c = mask_idx[s * MLEN + mi];
  float d = full[(size_t)s * HID + c * EE + j] - tokens[(size_t)s * HID + c * EE + j];
  sq_lat[idx] = d * d;
}

// deterministic final reduction -> loss
__global__ __launch_bounds__(256) void finish_kernel(const float* __restrict__ sq_sig,
                                                     const float* __restrict__ sq_lat,
                                                     float* __restrict__ loss_out) {
  __shared__ float red[256];
  int t = threadIdx.x;
  float a = 0.f, b = 0.f;
  for (int i = t; i < SS * LL; i += 256) a += sq_sig[i];
  for (int i = t; i < SS * MLEN * EE; i += 256) b += sq_lat[i];
  red[t] = a; __syncthreads();
  for (int st = 128; st > 0; st >>= 1) { if (t < st) red[t] += red[t + st]; __syncthreads(); }
  float ssig = red[0]; __syncthreads();
  red[t] = b; __syncthreads();
  for (int st = 128; st > 0; st >>= 1) { if (t < st) red[t] += red[t + st]; __syncthreads(); }
  if (t == 0)
    loss_out[0] = 0.7f * (ssig / (float)(SS * LL)) +
                  0.3f * (red[0] / (float)(SS * MLEN * EE));
}

// ------------------------------------------------------------------
// launch
// ------------------------------------------------------------------
#define CDIV(a, b) (((a) + (b) - 1) / (b))

extern "C" void kernel_launch(void* const* d_in, const int* in_sizes, int n_in,
                              void* d_out, int out_size, void* d_ws, size_t ws_size,
                              hipStream_t stream) {
  (void)in_sizes; (void)n_in; (void)out_size; (void)ws_size;
  const float* value = (const float*)d_in[0];
  const float* tts   = (const float*)d_in[1];
  const float* mask  = (const float*)d_in[2];
  const float* noise = (const float*)d_in[3];
  const float* tsw   = (const float*)d_in[4];
  const float* tsb   = (const float*)d_in[5];
  const float* tpw   = (const float*)d_in[6];
  const float* tpb   = (const float*)d_in[7];
  const float* fg1_w = (const float*)d_in[8];
  const float* fg1_b = (const float*)d_in[9];
  const float* fg2_w = (const float*)d_in[10];
  const float* fg2_b = (const float*)d_in[11];
  const float* fg3_w = (const float*)d_in[12];
  const float* fg3_b = (const float*)d_in[13];
  const float* t_bias = (const float*)d_in[14];
  const float* enc_in_w = (const float*)d_in[15];
  const float* enc_in_b = (const float*)d_in[16];
  const float* enc_out_w = (const float*)d_in[17];
  const float* enc_out_b = (const float*)d_in[18];
  const float* enc_ln1_g = (const float*)d_in[19];
  const float* enc_ln1_b = (const float*)d_in[20];
  const float* enc_ff1_w = (const float*)d_in[21];
  const float* enc_ff1_b = (const float*)d_in[22];
  const float* enc_ff2_w = (const float*)d_in[23];
  const float* enc_ff2_b = (const float*)d_in[24];
  const float* enc_ln2_g = (const float*)d_in[25];
  const float* enc_ln2_b = (const float*)d_in[26];
  const float* dec1_w = (const float*)d_in[27];
  const float* dec1_b = (const float*)d_in[28];
  const float* dec2_w = (const float*)d_in[29];
  const float* dec2_b = (const float*)d_in[30];
  const float* sp_w = (const float*)d_in[31];
  const float* sp_b = (const float*)d_in[32];

  float* loss_out = (float*)d_out;
  float* sig_out  = (float*)d_out + 1;
  float* outp     = (float*)d_out + 1 + BB * LL * DD;  // (B,D,HID) == (S,HID)

  // workspace carve-out
  char* wsb = (char*)d_ws;
  size_t off = 0;
  auto take = [&](size_t nbytes) -> void* {
    void* p = wsb + off;
    off += (nbytes + 255) & ~(size_t)255;
    return p;
  };
  float*    Xf32 = (float*)   take((size_t)SL * FINc * 4);
  _Float16* Xf16 = (_Float16*)take((size_t)SL * FPAD * 2);
  float*    mbuf = (float*)   take((size_t)SL * 4);
  _Float16* w1h  = (_Float16*)take((size_t)HID * FPAD * 2);
  _Float16* w2h  = (_Float16*)take((size_t)HID * HID * 2);
  _Float16* w3h  = (_Float16*)take((size_t)CTOT * HID * 2);
  _Float16* h1   = (_Float16*)take((size_t)SL * HID * 2);
  _Float16* h2   = (_Float16*)take((size_t)SL * HID * 2);
  float*    qkvb = (float*)   take((size_t)512 * 3 * EE * 4);
  float*    attnb= (float*)   take((size_t)512 * EE * 4);
  float*    hb   = (float*)   take((size_t)512 * 4 * EE * 4);
  float*    xb   = (float*)   take((size_t)512 * EE * 4);
  float*    db   = (float*)   take((size_t)512 * 4 * EE * 4);
  float*    dvb  = (float*)   take((size_t)512 * EE * 4);
  float*    zbar = (float*)   take((size_t)SS * EE * 4);
  float*    abar = (float*)   take((size_t)SS * 4 * EE * 4);
  float*    advb = (float*)   take((size_t)SS * EE * 4);
  float*    fullb= (float*)   take((size_t)SS * HID * 4);
  int*      vis_idx  = (int*) take((size_t)SS * NVIS * 4);
  int*      mask_idx = (int*) take((size_t)SS * MLEN * 4);
  float*    sq_sig = (float*) take((size_t)SS * LL * 4);
  float*    sq_lat = (float*) take((size_t)SS * MLEN * EE * 4);

  // 1) weight conversion to f16
  fg1_pad_kernel<<<CDIV(HID * FPAD, 256), 256, 0, stream>>>(fg1_w, w1h);
  f32_to_f16_kernel<<<CDIV(HID * HID, 256), 256, 0, stream>>>(fg2_w, w2h, HID * HID);
  f32_to_f16_kernel<<<CDIV(CTOT * HID, 256), 256, 0, stream>>>(fg3_w, w3h, CTOT * HID);

  // 2) features + mask + token permutation
  build_xf_kernel<<<CDIV(SL, 256), 256, 0, stream>>>(value, tts, mask, tsw, tsb,
                                                     tpw, tpb, Xf32, Xf16, mbuf);
  sort_kernel<<<1, 128, 0, stream>>>(noise, vis_idx, mask_idx);

  // 3) fg1, fg2 via WMMA: 32x32 tile/wave -> (8192/32)*(512/32)=4096 tiles,
  //    4 waves/block -> grid 1024
  gemm_wmma_kernel<<<(SL / 32) * (HID / 32) / 4, 128, 0, stream>>>(
      Xf16, w1h, fg1_b, h1, nullptr, SL, HID, FPAD, 1);
  gemm_wmma_kernel<<<(SL / 32) * (HID / 32) / 4, 128, 0, stream>>>(
      h1, w2h, fg2_b, h2, nullptr, SL, HID, HID, 1);

  // 4) fused fg3 + masked softmax + einsum + relu -> `output`
  fg3_fused_kernel<<<SS, 256, 0, stream>>>(h2, w3h, fg3_b, Xf32, mbuf, t_bias, outp);

  // 5) gather visible tokens -> encoder input (512,64)
  gather_vis_kernel<<<CDIV(SS * NVIS * EE, 256), 256, 0, stream>>>(outp, vis_idx, xb);

  // 6) 4 transformer encoder layers
  for (int i = 0; i < NLAYERS; ++i) {
    const float* iw = enc_in_w + (size_t)i * 3 * EE * EE;
    const float* ib = enc_in_b + (size_t)i * 3 * EE;
    const float* ow = enc_out_w + (size_t)i * EE * EE;
    const float* ob = enc_out_b + (size_t)i * EE;
    const float* g1 = enc_ln1_g + (size_t)i * EE;
    const float* b1 = enc_ln1_b + (size_t)i * EE;
    const float* f1w = enc_ff1_w + (size_t)i * 4 * EE * EE;
    const float* f1b = enc_ff1_b + (size_t)i * 4 * EE;
    const float* f2w = enc_ff2_w + (size_t)i * EE * 4 * EE;
    const float* f2b = enc_ff2_b + (size_t)i * EE;
    const float* g2 = enc_ln2_g + (size_t)i * EE;
    const float* b2 = enc_ln2_b + (size_t)i * EE;
    dense_kernel<<<CDIV(512 * 3 * EE, 256), 256, 0, stream>>>(xb, iw, ib, qkvb,
                                                              512, 3 * EE, EE, 0);
    attn_kernel<<<CDIV(NHEAD * 512, 256), 256, 0, stream>>>(qkvb, attnb);
    addproj_ln_kernel<<<512, 64, 0, stream>>>(xb, attnb, ow, ob, g1, b1, xb, EE);
    dense_kernel<<<CDIV(512 * 4 * EE, 256), 256, 0, stream>>>(xb, f1w, f1b, hb,
                                                              512, 4 * EE, EE, 2);
    addproj_ln_kernel<<<512, 64, 0, stream>>>(xb, hb, f2w, f2b, g2, b2, xb, 4 * EE);
  }

  // 7) decoder on visible tokens and on per-sample mean
  dense_kernel<<<CDIV(512 * 4 * EE, 256), 256, 0, stream>>>(xb, dec1_w, dec1_b, db,
                                                            512, 4 * EE, EE, 2);
  dense_kernel<<<CDIV(512 * EE, 256), 256, 0, stream>>>(db, dec2_w, dec2_b, dvb,
                                                        512, EE, 4 * EE, 0);
  avg4_kernel<<<CDIV(SS * EE, 256), 256, 0, stream>>>(xb, zbar);
  dense_kernel<<<CDIV(SS * 4 * EE, 256), 256, 0, stream>>>(zbar, dec1_w, dec1_b, abar,
                                                           SS, 4 * EE, EE, 2);
  dense_kernel<<<CDIV(SS * EE, 256), 256, 0, stream>>>(abar, dec2_w, dec2_b, advb,
                                                       SS, EE, 4 * EE, 0);

  // 8) assemble `full`, project signal, losses
  bcast_full_kernel<<<CDIV(SS * PP * EE, 256), 256, 0, stream>>>(advb, fullb);
  scatter_vis_kernel<<<CDIV(SS * NVIS * EE, 256), 256, 0, stream>>>(dvb, vis_idx, fullb);
  sig_kernel<<<CDIV(SS * LL, 256), 256, 0, stream>>>(fullb, sp_w, sp_b, value,
                                                     sig_out, sq_sig);
  latent_kernel<<<CDIV(SS * MLEN * EE, 256), 256, 0, stream>>>(fullb, outp, mask_idx,
                                                               sq_lat);
  finish_kernel<<<1, 256, 0, stream>>>(sq_sig, sq_lat, loss_out);
}
